// CVGA_8461085573268
// MI455X (gfx1250) — matrix operand order repaired and checked by reading, not compiled, not run on hardware
//
#include <hip/hip_runtime.h>
#include <hip/hip_bf16.h>

typedef __attribute__((ext_vector_type(16))) __bf16 v16bf;
typedef __attribute__((ext_vector_type(8)))  float  v8f;
typedef __attribute__((ext_vector_type(4)))  float  f4;

#define N_USERS   50000
#define N_ITEMS   40000
#define EMB_D     64
#define H_DIM     128
#define NNZ_E     2000000
#define BATCH_B   1024
#define NCHUNK    100          // 40000 cols / 400 per wave-chunk
#define CHUNK_C   400

// ---- bf16 split helpers (truncating hi, residual lo) ----
__device__ __forceinline__ __bf16 f2bf(float f) {
    unsigned u = __builtin_bit_cast(unsigned, f);
    unsigned short h = (unsigned short)(u >> 16);
    return __builtin_bit_cast(__bf16, h);
}
__device__ __forceinline__ float bfhi(float f) {
    unsigned u = __builtin_bit_cast(unsigned, f) & 0xFFFF0000u;
    return __builtin_bit_cast(float, u);
}

// fragment view: 16 bf16 = two 16B vectors
union BFrag { v16bf v; f4 q[2]; };

// ---- 1. zero the two loss accumulators ----
__global__ void k_zero(float* acc) {
    if (threadIdx.x < 2) acc[threadIdx.x] = 0.0f;
}

// ---- 2. h = bq (broadcast bias init before scatter) ----
__global__ void __launch_bounds__(256) k_inith(const float* __restrict__ bq,
                                               float* __restrict__ h) {
    long long i = (long long)blockIdx.x * 256 + threadIdx.x;
    if (i < (long long)N_USERS * H_DIM) h[i] = bq[(int)(i & (H_DIM - 1))];
}

// ---- 3. WqT[col][c] = Wq[c][col]  (coalesced reads) ----
__global__ void __launch_bounds__(256) k_transpose(const float* __restrict__ Wq,
                                                   float* __restrict__ WqT) {
    int idx = blockIdx.x * 256 + threadIdx.x;
    if (idx >= N_ITEMS * H_DIM) return;
    int col = idx % N_ITEMS;
    int c   = idx / N_ITEMS;
    WqT[(size_t)col * H_DIM + c] = Wq[idx];
}

// ---- 3b. one-shot bf16 hi/lo split of Wp (loop-invariant across row tiles) ----
__global__ void __launch_bounds__(256) k_prepw(const float* __restrict__ Wp,
                                               __bf16* __restrict__ Wph,
                                               __bf16* __restrict__ Wpl) {
    int idx = blockIdx.x * 256 + threadIdx.x;
    if (idx >= N_ITEMS * EMB_D) return;
    float f = Wp[idx];
    Wph[idx] = f2bf(f);
    Wpl[idx] = f2bf(f - bfhi(f));
}

// ---- 4. SpMM scatter: one wave per edge, 128 channels / 32 lanes ----
__global__ void __launch_bounds__(256) k_spmm(const int* __restrict__ rows,
                                              const int* __restrict__ cols,
                                              const float* __restrict__ vals,
                                              const float* __restrict__ WqT,
                                              float* __restrict__ h) {
    long long gt = (long long)blockIdx.x * 256 + threadIdx.x;
    int e = (int)(gt >> 5);
    if (e >= NNZ_E) return;
    int lane = threadIdx.x & 31;
    int r = rows[e];
    int c = cols[e];
    float v = vals[e];
    const f4* w = (const f4*)(WqT + (size_t)c * H_DIM);
    f4 wv = w[lane];                     // contiguous 512B per edge
    float* hp = h + (size_t)r * H_DIM + lane * 4;
    atomicAdd(hp + 0, v * wv[0]);
    atomicAdd(hp + 1, v * wv[1]);
    atomicAdd(hp + 2, v * wv[2]);
    atomicAdd(hp + 3, v * wv[3]);
}

// ---- 5. z = mu + eps*exp(0.5*logvar) for batch users; KL partials ----
__global__ void __launch_bounds__(64) k_zkl(const float* __restrict__ h,
                                            const float* __restrict__ eps,
                                            const int* __restrict__ user,
                                            float* __restrict__ zb,
                                            float* __restrict__ acc) {
    int b = blockIdx.x;
    int d = threadIdx.x;               // 0..63
    int u = user[b];
    float mu = h[(size_t)u * H_DIM + d];
    float lv = h[(size_t)u * H_DIM + EMB_D + d];
    float z  = mu + eps[(size_t)u * EMB_D + d] * __expf(0.5f * lv);
    zb[b * EMB_D + d] = z;
    float kt = 1.0f + lv - mu * mu - __expf(lv);
    for (int off = 16; off; off >>= 1) kt += __shfl_down(kt, off, 32);
    if ((threadIdx.x & 31) == 0) atomicAdd(&acc[1], kt);
}

// ---- 6. fused bf16x3-WMMA GEMM + online logsumexp + x-masked sum ----
// grid (25, 64), block 128 (4 waves). Wave w of block bx owns columns
// [ (bx*4+w)*400 , +400 ) for row tile blockIdx.y*16.
__global__ void __launch_bounds__(128) k_gemm(const float* __restrict__ zb,
                                              const __bf16* __restrict__ Wph,
                                              const __bf16* __restrict__ Wpl,
                                              const float* __restrict__ bp,
                                              const float* __restrict__ x,
                                              float* __restrict__ pm,
                                              float* __restrict__ ps,
                                              float* __restrict__ psxr,
                                              float* __restrict__ pnx) {
    const int lane = threadIdx.x & 31;
    const int wave = threadIdx.x >> 5;
    const int half = lane >> 4;         // 0 or 1
    const int nn   = lane & 15;
    const int rowBase = blockIdx.y * 16;
    const int chunk   = blockIdx.x * 4 + wave;
    const int colBase = chunk * CHUNK_C;

    // A fragments (16x32 bf16 layout): lane holds M = nn,
    // K(i) = s*32 + half*16 + i. Converted once per wave.
    v16bf ah[2], al[2];
    const float* arow = zb + (size_t)(rowBase + nn) * EMB_D;
    #pragma unroll
    for (int s = 0; s < 2; ++s) {
        #pragma unroll
        for (int i = 0; i < 16; ++i) {
            int K = s * 32 + half * 16 + i;
            float f = arow[K];
            ah[s][i] = f2bf(f);
            al[s][i] = f2bf(f - bfhi(f));
        }
    }

    // per-lane online stats for 8 rows (C-layout slots)
    float m8[8], s8[8], sxr8[8], nx8[8];
    #pragma unroll
    for (int r = 0; r < 8; ++r) { m8[r] = -3.0e38f; s8[r] = 0.f; sxr8[r] = 0.f; nx8[r] = 0.f; }

    for (int t = 0; t < CHUNK_C / 16; ++t) {
        int j0 = colBase + t * 16;
        v8f c = {};
        #pragma unroll
        for (int s = 0; s < 2; ++s) {
            // B fragment (32x16 bf16): lane holds N = nn, K = s*32 + half*16 + i
            // 32B contiguous per lane; pre-split to bf16 -> pure b128 loads, no VALU.
            size_t boff = (size_t)(j0 + nn) * EMB_D + s * 32 + half * 16;
            const f4* ph = (const f4*)(Wph + boff);
            const f4* pl = (const f4*)(Wpl + boff);
            BFrag bh, bl;
            bh.q[0] = ph[0]; bh.q[1] = ph[1];
            bl.q[0] = pl[0]; bl.q[1] = pl[1];
            // bf16x3 split: hi*hi + hi*lo + lo*hi ~ fp32 accuracy
            c = __builtin_amdgcn_wmma_f32_16x16x32_bf16(false, ah[s], false, bh.v, (short)0, c, false, false);
            c = __builtin_amdgcn_wmma_f32_16x16x32_bf16(false, ah[s], false, bl.v, (short)0, c, false, false);
            c = __builtin_amdgcn_wmma_f32_16x16x32_bf16(false, al[s], false, bh.v, (short)0, c, false, false);
        }
        float bias = bp[j0 + nn];
        #pragma unroll
        for (int r = 0; r < 8; ++r) {
            float v = c[r] + bias;                       // recon[row, j0+nn]
            int row = rowBase + r + 8 * half;
            float xv = x[(size_t)row * N_ITEMS + j0 + nn];
            float nm = fmaxf(m8[r], v);
            s8[r] = s8[r] * __expf(m8[r] - nm) + __expf(v - nm);
            m8[r] = nm;
            sxr8[r] += xv * v;
            nx8[r]  += xv;
        }
    }

    // merge the 16 lanes sharing each row (xor masks stay inside each half)
    #pragma unroll
    for (int r = 0; r < 8; ++r) {
        float m = m8[r], s = s8[r], sx = sxr8[r], nx = nx8[r];
        #pragma unroll
        for (int mask = 1; mask < 16; mask <<= 1) {
            float mo = __shfl_xor(m, mask, 32);
            float so = __shfl_xor(s, mask, 32);
            float nm = fmaxf(m, mo);
            s = s * __expf(m - nm) + so * __expf(mo - nm);
            m = nm;
            sx += __shfl_xor(sx, mask, 32);
            nx += __shfl_xor(nx, mask, 32);
        }
        if (nn == 0) {
            int row = rowBase + r + 8 * half;
            size_t p = (size_t)row * NCHUNK + chunk;
            pm[p] = m; ps[p] = s; psxr[p] = sx; pnx[p] = nx;
        }
    }
}

// ---- 7. merge 100 chunk partials per row -> recon loss accumulator ----
__global__ void __launch_bounds__(256) k_reduce(const float* __restrict__ pm,
                                                const float* __restrict__ ps,
                                                const float* __restrict__ psxr,
                                                const float* __restrict__ pnx,
                                                float* __restrict__ acc) {
    int row = blockIdx.x * 256 + threadIdx.x;
    if (row >= BATCH_B) return;
    float M = -3.0e38f, S = 0.f, sx = 0.f, nx = 0.f;
    for (int c = 0; c < NCHUNK; ++c) {
        size_t p = (size_t)row * NCHUNK + c;
        float m2 = pm[p], s2 = ps[p];
        float nm = fmaxf(M, m2);
        S = S * __expf(M - nm) + s2 * __expf(m2 - nm);
        M = nm;
        sx += psxr[p];
        nx += pnx[p];
    }
    float lse = M + __logf(S);
    // -sum(x*(recon - lse)) = nx*lse - sum(x*recon)
    atomicAdd(&acc[0], nx * lse - sx);
}

// ---- 8. finalize the two scalars ----
__global__ void k_final(const float* __restrict__ acc, float* __restrict__ out) {
    if (threadIdx.x == 0) {
        out[0] = acc[0] / (float)BATCH_B;
        out[1] = -0.5f * acc[1] / (float)BATCH_B;
    }
}

extern "C" void kernel_launch(void* const* d_in, const int* in_sizes, int n_in,
                              void* d_out, int out_size, void* d_ws, size_t ws_size,
                              hipStream_t stream) {
    const float* graph_vals = (const float*)d_in[0];
    const float* Wq         = (const float*)d_in[1];
    const float* bq         = (const float*)d_in[2];
    const float* Wp         = (const float*)d_in[3];
    const float* bp         = (const float*)d_in[4];
    const float* x          = (const float*)d_in[5];
    const float* eps        = (const float*)d_in[6];
    const int*   graph_rows = (const int*)d_in[7];
    const int*   graph_cols = (const int*)d_in[8];
    const int*   user       = (const int*)d_in[9];
    float* out = (float*)d_out;

    // workspace layout (floats, then bf16 arrays; all offsets keep 32B alignment)
    float* ws   = (float*)d_ws;
    float* h    = ws;                                   // 50000*128 = 6,400,000
    float* WqT  = h    + (size_t)N_USERS * H_DIM;       // 40000*128 = 5,120,000
    float* zb   = WqT  + (size_t)N_ITEMS * H_DIM;       // 1024*64   = 65,536
    float* pm   = zb   + (size_t)BATCH_B * EMB_D;       // 1024*100  = 102,400
    float* ps   = pm   + (size_t)BATCH_B * NCHUNK;
    float* psxr = ps   + (size_t)BATCH_B * NCHUNK;
    float* pnx  = psxr + (size_t)BATCH_B * NCHUNK;
    float* acc  = pnx  + (size_t)BATCH_B * NCHUNK;      // 2 floats (padded to 8)
    __bf16* Wph = (__bf16*)(acc + 8);                   // 2,560,000 bf16
    __bf16* Wpl = Wph + (size_t)N_ITEMS * EMB_D;        // 2,560,000 bf16

    k_zero<<<1, 64, 0, stream>>>(acc);
    k_inith<<<(N_USERS * H_DIM) / 256, 256, 0, stream>>>(bq, h);
    k_transpose<<<(N_ITEMS * H_DIM) / 256, 256, 0, stream>>>(Wq, WqT);
    k_prepw<<<(N_ITEMS * EMB_D) / 256, 256, 0, stream>>>(Wp, Wph, Wpl);
    k_spmm<<<(NNZ_E * 32) / 256, 256, 0, stream>>>(graph_rows, graph_cols,
                                                   graph_vals, WqT, h);
    k_zkl<<<BATCH_B, 64, 0, stream>>>(h, eps, user, zb, acc);
    k_gemm<<<dim3(25, BATCH_B / 16), 128, 0, stream>>>(zb, Wph, Wpl, bp, x,
                                                       pm, ps, psxr, pnx);
    k_reduce<<<BATCH_B / 256, 256, 0, stream>>>(pm, ps, psxr, pnx, acc);
    k_final<<<1, 32, 0, stream>>>(acc, out);
}
